// MultiHeadAttentionQuantum_65481071409102
// MI455X (gfx1250) — compile-verified, matrix operand order checked by reading, not run on hardware
//
#include <hip/hip_runtime.h>
#include <hip/hip_bf16.h>

typedef __attribute__((ext_vector_type(16))) _Float16 v16h;
typedef __attribute__((ext_vector_type(8)))  float    v8f;

#define E_DIM 512
#define MROWS 65536          // B*S = 8*8192
#define KSTEPS 16            // 512 / 32
#define NTILES 32            // 512 / 16

// ---------------------------------------------------------------------------
// Prepack W (f32, [f,e] row-major) into f16 WMMA B-fragments (32x16, K x N).
// Fragment storage (kt-major so one wave's per-K-step fragments are at a
// single base + small immediate offsets):
//   dword index = ((kt*NTILES + nt)*32 + lane)*8 + d
// Per ISA 7.12.2 (16-bit B layout, wave32):
//   lane 0-15  -> N = nt*16 + lane,    K-half h=0
//   lane 16-31 -> N = nt*16 + lane-16, K-half h=1
//   dword d<4  -> k_local = h*8 + 2d      (pair k, k+1)
//   dword d>=4 -> k_local = 16 + h*8 + 2(d-4)
// B[k][n] = W[n*E + k]  (out = meas @ W^T).
// ---------------------------------------------------------------------------
__global__ __launch_bounds__(256) void qmha_prepack_w(
    const float* __restrict__ W, _Float16* __restrict__ Bp)
{
    int tid = blockIdx.x * 256 + threadIdx.x;      // one thread per f16 pair
    if (tid >= KSTEPS * NTILES * 32 * 8) return;
    int d    = tid & 7;
    int lane = (tid >> 3) & 31;
    int nt   = (tid >> 8) & (NTILES - 1);
    int kt   = tid >> 13;

    int h  = lane >> 4;
    int n  = nt * 16 + (lane & 15);
    int kl = (d < 4) ? (h * 8 + 2 * d) : (16 + h * 8 + 2 * (d - 4));
    int k  = kt * 32 + kl;

    Bp[(size_t)tid * 2 + 0] = (_Float16)W[(size_t)n * E_DIM + k];
    Bp[(size_t)tid * 2 + 1] = (_Float16)W[(size_t)n * E_DIM + k + 1];
}

// ---------------------------------------------------------------------------
// Fused cos + GEMM: A = cos(x + theta) built on the fly in registers (f16),
// v_wmma_f32_16x16x32_f16 with f32 accumulation, f32 non-temporal stores.
// Block = 256 threads = 8 waves; block tile = 32 rows x 512 cols.
//   wave>>2 -> 16-row half; wave&3 -> 128-col (8 n-tile) slab.
// B fragments are 2-deep software-pipelined so one fragment load is in
// flight under each WMMA.
// ---------------------------------------------------------------------------
__global__ __launch_bounds__(256) void qmha_cos_gemm(
    const float* __restrict__ x, const float* __restrict__ theta,
    const _Float16* __restrict__ Bp, float* __restrict__ out)
{
    const int lane   = threadIdx.x & 31;
    const int wave   = threadIdx.x >> 5;
    const int mBase  = blockIdx.x * 32 + (wave >> 2) * 16;
    const int ntBase = (wave & 3) * 8;
    const int h      = lane >> 4;
    const int lm     = lane & 15;

    float th[8];
#pragma unroll
    for (int i = 0; i < 8; ++i) th[i] = theta[i];

    // x: per-lane fixed base; per K-step advance is a constant immediate.
    // Lane reads its K-half contiguously: k_local h*8..h*8+7 and 16+h*8..+7.
    const char* xbase = (const char*)x
                      + ((size_t)(mBase + lm) * E_DIM + (size_t)h * 8) * 4;
    // B: uniform base (ntBase) + lane*32 voffset; per K-step +32768 uniform,
    // per n-tile +1024 immediate.
    const char* Bbase = (const char*)Bp + (size_t)ntBase * 1024 + (size_t)lane * 32;

    v8f acc[8] = {};

    for (int kt = 0; kt < KSTEPS; ++kt) {
        const char* xk = xbase + kt * 128;
        const float4 x0 = *(const float4*)(xk +  0);   // k_local h*8+0..3
        const float4 x1 = *(const float4*)(xk + 16);   // k_local h*8+4..7
        const float4 x2 = *(const float4*)(xk + 64);   // k_local 16+h*8+0..3
        const float4 x3 = *(const float4*)(xk + 80);   // k_local 16+h*8+4..7

        const char* Bk = Bbase + kt * (NTILES * 32 * 32);
        v16h bcur = *(const v16h*)(Bk);                // fragment t=0 in flight

        float xv[16] = { x0.x, x0.y, x0.z, x0.w,  x1.x, x1.y, x1.z, x1.w,
                         x2.x, x2.y, x2.z, x2.w,  x3.x, x3.y, x3.z, x3.w };
        v16h a;
#pragma unroll
        for (int i = 0; i < 16; ++i)
            a[i] = (_Float16)__cosf(xv[i] + th[i & 7]);   // theta idx = k % 8

#pragma unroll
        for (int t = 0; t < 8; ++t) {
            v16h bnext = bcur;
            if (t < 7) bnext = *(const v16h*)(Bk + (t + 1) * 1024);
            acc[t] = __builtin_amdgcn_wmma_f32_16x16x32_f16(
                false, a, false, bcur, (short)0, acc[t], false, false);
            bcur = bnext;
        }
    }

    // Store D: VGPR r -> row mBase + r + h*8, col (ntBase+t)*16 + lm.
#pragma unroll
    for (int t = 0; t < 8; ++t) {
        const int n = (ntBase + t) * 16 + lm;
#pragma unroll
        for (int r = 0; r < 8; ++r) {
            const int m = mBase + r + h * 8;
            __builtin_nontemporal_store(acc[t][r], &out[(size_t)m * E_DIM + n]);
        }
    }
}

extern "C" void kernel_launch(void* const* d_in, const int* in_sizes, int n_in,
                              void* d_out, int out_size, void* d_ws, size_t ws_size,
                              hipStream_t stream) {
    const float* x     = (const float*)d_in[0];   // [8, 8192, 512] f32
    const float* theta = (const float*)d_in[1];   // [8] f32
    const float* W     = (const float*)d_in[2];   // [512, 512] f32
    float*       out   = (float*)d_out;           // [8, 8192, 512] f32
    _Float16*    Bp    = (_Float16*)d_ws;         // 512*512 f16 = 512 KB

    const int total = KSTEPS * NTILES * 32 * 8;   // 131072 f16-pairs
    qmha_prepack_w<<<(total + 255) / 256, 256, 0, stream>>>(W, Bp);
    qmha_cos_gemm<<<MROWS / 32, 256, 0, stream>>>(x, theta, Bp, out);
}
